// MPNN_edge_sparse_63780264346297
// MI455X (gfx1250) — matrix operand order repaired and checked by reading, not compiled, not run on hardware
//
#include <hip/hip_runtime.h>

typedef __attribute__((ext_vector_type(16))) _Float16 v16h;
typedef __attribute__((ext_vector_type(8)))  _Float16 v8h;
typedef __attribute__((ext_vector_type(8)))  float    v8f;

#define WPB 8            // waves per block (256 threads)
// Prepped f16 weight-fragment region (offsets in halfs), one 32x16 B-frag = 512 halfs:
//   W1m (kt 0..2, nt 0..3):      (kt*4+nt)*512          [6144 halfs]
//   W2m (kt 0..1, nt 0..1): 6144+(kt*2+nt)*512          [2048]
//   W1u (kt 0..1, nt 0..3): 8192+(kt*4+nt)*512          [4096]
//   W2u (kt 0..1, nt 0..1):12288+(kt*2+nt)*512          [2048]
#define WF_TOTAL 14336
#define WF_EDGE   8192
#define WF_NODE   6144

static __device__ __forceinline__ float4 ld4(const float* p) {
  return *(const float4*)p;
}

// A-fragment (16-bit A, 16x32): lane holds row M=lane%16; with g=lane/16, b=8g the
// 16 halfs are features [b..b+7] then [16+b..16+b+7] of the 32-wide K tile.
static __device__ __forceinline__ v16h packA(float4 a, float4 b, float4 c, float4 d) {
  v16h r;
  r[0]=(_Float16)a.x;  r[1]=(_Float16)a.y;  r[2]=(_Float16)a.z;  r[3]=(_Float16)a.w;
  r[4]=(_Float16)b.x;  r[5]=(_Float16)b.y;  r[6]=(_Float16)b.z;  r[7]=(_Float16)b.w;
  r[8]=(_Float16)c.x;  r[9]=(_Float16)c.y;  r[10]=(_Float16)c.z; r[11]=(_Float16)c.w;
  r[12]=(_Float16)d.x; r[13]=(_Float16)d.y; r[14]=(_Float16)d.z; r[15]=(_Float16)d.w;
  return r;
}

// edge-feature tile: 8 real halfs, upper 8 are K-padding (K=80..95 -> 0)
static __device__ __forceinline__ v16h packEF(float4 a, float4 b) {
  v16h r;
  r[0]=(_Float16)a.x; r[1]=(_Float16)a.y; r[2]=(_Float16)a.z; r[3]=(_Float16)a.w;
  r[4]=(_Float16)b.x; r[5]=(_Float16)b.y; r[6]=(_Float16)b.z; r[7]=(_Float16)b.w;
  _Float16 z = (_Float16)0.0f;
  r[8]=z; r[9]=z; r[10]=z; r[11]=z; r[12]=z; r[13]=z; r[14]=z; r[15]=z;
  return r;
}

static __device__ __forceinline__ v16h cat8(v8h lo, v8h hi) {
  return __builtin_shufflevector(lo, hi, 0,1,2,3,4,5,6,7,8,9,10,11,12,13,14,15);
}

static __device__ __forceinline__ v16h ldB(const _Float16* base, int lane) {
  return *(const v16h*)(base + lane * 16);   // 32B per lane, LDS, 2x ds_load_b128
}

static __device__ __forceinline__ v8f wmma_f16(v16h a, v16h b, v8f c) {
  // D = A(16x32 f16) x B(32x16 f16) + C(16x16 f32)
  return __builtin_amdgcn_wmma_f32_16x16x32_f16(false, a, false, b, (short)0, c,
                                                false, false);
}

static __device__ __forceinline__ v8f splat8(float v) {
  v8f c;
  for (int i = 0; i < 8; ++i) c[i] = v;
  return c;
}

__global__ __launch_bounds__(256) void mpnn_zero_f32(float* __restrict__ p, int n) {
  int i = blockIdx.x * 256 + threadIdx.x;
  if (i < n) p[i] = 0.0f;
}

// Convert fp32 row-major weights into f16 WMMA B-fragments.
// B-fragment (16-bit B, 32x16): lane holds col N=lane%16; half s=2v+h holds
// K = kt*32 + (lane/16)*8 + (s/8)*16 + s%8. K >= Kreal is zero padding.
__global__ __launch_bounds__(256) void mpnn_prep_weights(
    const float* __restrict__ W1m, const float* __restrict__ W2m,
    const float* __restrict__ W1u, const float* __restrict__ W2u,
    _Float16* __restrict__ out)
{
  int t = blockIdx.x * 256 + threadIdx.x;      // 28 frags x 32 lanes = 896 entries
  if (t >= 896) return;
  int lane = t & 31;
  int frag = t >> 5;

  const float* W; int Kreal, Ncols, kt, nt, off;
  if (frag < 12)      { int f = frag;      W = W1m; Kreal = 80; Ncols = 64; kt = f/4; nt = f%4; off = 0     + f*512; }
  else if (frag < 16) { int f = frag - 12; W = W2m; Kreal = 64; Ncols = 32; kt = f/2; nt = f%2; off = 6144  + f*512; }
  else if (frag < 24) { int f = frag - 16; W = W1u; Kreal = 64; Ncols = 64; kt = f/4; nt = f%4; off = 8192  + f*512; }
  else                { int f = frag - 24; W = W2u; Kreal = 64; Ncols = 32; kt = f/2; nt = f%2; off = 12288 + f*512; }

  int m = lane & 15, g = lane >> 4;
  int N = nt * 16 + m;
  _Float16* o = out + off + lane * 16;
  for (int v = 0; v < 8; ++v)
    for (int h = 0; h < 2; ++h) {
      int K = kt * 32 + g * 8 + (v >> 2) * 16 + ((2 * v + h) & 7);
      float val = (K < Kreal) ? W[(size_t)K * Ncols + N] : 0.0f;
      o[2 * v + h] = (_Float16)val;
    }
}

// Edge kernel: one wave computes the 2-layer message MLP for 16 edges with WMMA,
// then scatters the 16x32 message tile into agg with f32 atomics (L2 resident).
__global__ __launch_bounds__(256) void mpnn_edge_kernel(
    const float* __restrict__ x, const float* __restrict__ ef,
    const int* __restrict__ eidx, const _Float16* __restrict__ wfr,
    const float* __restrict__ b1m, const float* __restrict__ b2m,
    float* __restrict__ agg, int E, int NT)
{
  __shared__ __align__(32) _Float16 wlds[WF_EDGE];
  __shared__ __align__(32) _Float16 hbuf[WPB][16][64];
  __shared__ int dsti[WPB][16];

  for (int i = threadIdx.x; i < WF_EDGE; i += 256) wlds[i] = wfr[i];
  __syncthreads();

  const int wave = threadIdx.x >> 5;
  const int lane = threadIdx.x & 31;
  const int m = lane & 15;
  const int g = lane >> 4;
  const int b = g * 8;
  const int tile = blockIdx.x * WPB + wave;
  if (tile >= NT) return;                      // wave-uniform exit; EXEC stays all-1s

  long long e = (long long)tile * 16 + m;
  long long ec = e < (long long)E ? e : (long long)E - 1;
  const int srcN = eidx[ec];
  const int dstN = eidx[(long long)E + ec];
  dsti[wave][m] = dstN;                        // lanes 0-15 and 16-31 write same value

  const float* xi  = x  + (size_t)dstN * 32;
  const float* xj  = x  + (size_t)srcN * 32;
  const float* efe = ef + (size_t)ec * 16;

  // A fragments for K tiles: [x_i | x_j | ef,pad]
  v16h a0 = packA(ld4(xi + b),  ld4(xi + b + 4),  ld4(xi + 16 + b),  ld4(xi + 20 + b));
  v16h a1 = packA(ld4(xj + b),  ld4(xj + b + 4),  ld4(xj + 16 + b),  ld4(xj + 20 + b));
  v16h a2 = packEF(ld4(efe + b), ld4(efe + b + 4));

  // Layer 1: [16x96] x [96x64] + b1m, ReLU; D tiles -> LDS for re-layout
  for (int nt = 0; nt < 4; ++nt) {
    v8f c = splat8(b1m[nt * 16 + m]);
    c = wmma_f16(a0, ldB(wlds + (0 * 4 + nt) * 512, lane), c);
    c = wmma_f16(a1, ldB(wlds + (1 * 4 + nt) * 512, lane), c);
    c = wmma_f16(a2, ldB(wlds + (2 * 4 + nt) * 512, lane), c);
    for (int r = 0; r < 8; ++r) {
      float v = c[r] > 0.0f ? c[r] : 0.0f;
      hbuf[wave][r + 8 * g][nt * 16 + m] = (_Float16)v;
    }
  }

  // Re-layout D(row-major in LDS) -> A fragments for layer 2 (K=64)
  const _Float16* hr = &hbuf[wave][m][0];
  v16h u0 = cat8(*(const v8h*)(hr + b),      *(const v8h*)(hr + 16 + b));
  v16h u1 = cat8(*(const v8h*)(hr + 32 + b), *(const v8h*)(hr + 48 + b));

  // Layer 2: [16x64] x [64x32] + b2m, then atomic scatter-add by destination
  for (int nt = 0; nt < 2; ++nt) {
    v8f c = splat8(b2m[nt * 16 + m]);
    c = wmma_f16(u0, ldB(wlds + 6144 + (0 * 2 + nt) * 512, lane), c);
    c = wmma_f16(u1, ldB(wlds + 6144 + (1 * 2 + nt) * 512, lane), c);
    for (int r = 0; r < 8; ++r) {
      int M = r + 8 * g;
      if ((long long)tile * 16 + M < (long long)E) {
        atomicAdd(&agg[(size_t)dsti[wave][M] * 32 + nt * 16 + m], c[r]);
      }
    }
  }
}

// Node kernel: 2-layer update MLP on [x | agg] per 16-node tile.
__global__ __launch_bounds__(256) void mpnn_node_kernel(
    const float* __restrict__ x, const float* __restrict__ agg,
    const _Float16* __restrict__ wfr,
    const float* __restrict__ b1u, const float* __restrict__ b2u,
    float* __restrict__ out, int N, int NT)
{
  __shared__ __align__(32) _Float16 wlds[WF_NODE];
  __shared__ __align__(32) _Float16 hbuf[WPB][16][64];

  for (int i = threadIdx.x; i < WF_NODE; i += 256) wlds[i] = wfr[8192 + i];
  __syncthreads();

  const int wave = threadIdx.x >> 5;
  const int lane = threadIdx.x & 31;
  const int m = lane & 15;
  const int g = lane >> 4;
  const int b = g * 8;
  const int tile = blockIdx.x * WPB + wave;
  if (tile >= NT) return;

  int n = tile * 16 + m;
  int nc = n < N ? n : N - 1;
  const float* xr = x   + (size_t)nc * 32;
  const float* ar = agg + (size_t)nc * 32;

  v16h a0 = packA(ld4(xr + b), ld4(xr + b + 4), ld4(xr + 16 + b), ld4(xr + 20 + b));
  v16h a1 = packA(ld4(ar + b), ld4(ar + b + 4), ld4(ar + 16 + b), ld4(ar + 20 + b));

  for (int nt = 0; nt < 4; ++nt) {
    v8f c = splat8(b1u[nt * 16 + m]);
    c = wmma_f16(a0, ldB(wlds + (0 * 4 + nt) * 512, lane), c);
    c = wmma_f16(a1, ldB(wlds + (1 * 4 + nt) * 512, lane), c);
    for (int r = 0; r < 8; ++r) {
      float v = c[r] > 0.0f ? c[r] : 0.0f;
      hbuf[wave][r + 8 * g][nt * 16 + m] = (_Float16)v;
    }
  }

  const _Float16* hr = &hbuf[wave][m][0];
  v16h u0 = cat8(*(const v8h*)(hr + b),      *(const v8h*)(hr + 16 + b));
  v16h u1 = cat8(*(const v8h*)(hr + 32 + b), *(const v8h*)(hr + 48 + b));

  for (int nt = 0; nt < 2; ++nt) {
    v8f c = splat8(b2u[nt * 16 + m]);
    c = wmma_f16(u0, ldB(wlds + 4096 + (0 * 2 + nt) * 512, lane), c);
    c = wmma_f16(u1, ldB(wlds + 4096 + (1 * 2 + nt) * 512, lane), c);
    for (int r = 0; r < 8; ++r) {
      int M = r + 8 * g;
      int node = tile * 16 + M;
      if (node < N) out[(size_t)node * 32 + nt * 16 + m] = c[r];
    }
  }
}

extern "C" void kernel_launch(void* const* d_in, const int* in_sizes, int n_in,
                              void* d_out, int out_size, void* d_ws, size_t ws_size,
                              hipStream_t stream) {
  (void)n_in; (void)out_size; (void)ws_size;
  const float* x    = (const float*)d_in[0];
  // d_in[1] = degrees (unused by reference)
  const float* ef   = (const float*)d_in[2];
  const float* W1m  = (const float*)d_in[3];
  const float* b1m  = (const float*)d_in[4];
  const float* W2m  = (const float*)d_in[5];
  const float* b2m  = (const float*)d_in[6];
  const float* W1u  = (const float*)d_in[7];
  const float* b1u  = (const float*)d_in[8];
  const float* W2u  = (const float*)d_in[9];
  const float* b2u  = (const float*)d_in[10];
  const int*   eidx = (const int*)d_in[11];

  const int N = in_sizes[0] / 32;   // nodes
  const int E = in_sizes[2] / 16;   // edges

  float* agg = (float*)d_ws;                                      // N*32 f32
  _Float16* wfr = (_Float16*)((char*)d_ws + (size_t)N * 32 * 4);  // WF_TOTAL halfs

  {
    int n = N * 32;
    mpnn_zero_f32<<<(n + 255) / 256, 256, 0, stream>>>(agg, n);
  }
  mpnn_prep_weights<<<4, 256, 0, stream>>>(W1m, W2m, W1u, W2u, wfr);
  {
    int NTe = (E + 15) / 16;
    mpnn_edge_kernel<<<(NTe + WPB - 1) / WPB, 256, 0, stream>>>(
        x, ef, eidx, wfr, b1m, b2m, agg, E, NTe);
  }
  {
    int NTn = (N + 15) / 16;
    mpnn_node_kernel<<<(NTn + WPB - 1) / WPB, 256, 0, stream>>>(
        x, agg, wfr, b1u, b2u, (float*)d_out, N, NTn);
  }
}
